// VVRWKV_SpatialMix_23373212025547
// MI455X (gfx1250) — matrix operand order repaired and checked by reading, not compiled, not run on hardware
//
#include <hip/hip_runtime.h>
#include <hip/hip_bf16.h>
#include <stdint.h>

typedef __attribute__((ext_vector_type(16))) __bf16 v16bf;
typedef __attribute__((ext_vector_type(8)))  __bf16 v8bf;
typedef __attribute__((ext_vector_type(8)))  float  v8f;
typedef uint32_t u32x4 __attribute__((ext_vector_type(4)));
typedef uint32_t u32x8 __attribute__((ext_vector_type(8)));

#define BB 8
#define TT 1024
#define CC 768
#define MM (BB*TT)            // 8192 rows
#define PL ((size_t)MM*CC)    // one f32 plane, elements
#define NCHAN (BB*4*CC)       // 24576 scan channels
#define NCH 16                // scan chunks
#define CHL (TT/NCH)          // 64 steps per chunk

__device__ __forceinline__ unsigned short f2bf(float f) {
  union { float f; unsigned u; } v; v.f = f;
  unsigned r = v.u + 0x7FFFu + ((v.u >> 16) & 1u);   // round-to-nearest-even
  return (unsigned short)(r >> 16);
}

__device__ __forceinline__ int scan_t(int dir, int s) {
  if (dir == 0) return s;
  if (dir == 1) return ((s & 31) << 5) | (s >> 5);
  if (dir == 2) return (TT - 1) - s;
  return ((31 - (s & 31)) << 5) | (31 - (s >> 5));
}

// ---------------------------------------------------------------------------
// Kernel 1: q_shift + token mixing, emit bf16 GEMM inputs xk/xv/xr
// ---------------------------------------------------------------------------
__global__ __launch_bounds__(256) void qshift_mix(
    const float* __restrict__ x,
    const float* __restrict__ mk, const float* __restrict__ mv,
    const float* __restrict__ mr,
    unsigned short* __restrict__ xkb, unsigned short* __restrict__ xvb,
    unsigned short* __restrict__ xrb)
{
  int i = blockIdx.x * 256 + threadIdx.x;          // over B*T*C
  int c = i % CC;
  int t = (i / CC) & (TT - 1);
  int xc = t & 31, yr = (t >> 5) & 31;
  float xv_ = x[i];
  float xx = 0.f;
  int grp = c / (CC / 4);
  if (grp == 0)      { if (xc > 0)  xx = x[i - CC]; }        // shift right
  else if (grp == 1) { if (xc < 31) xx = x[i + CC]; }        // shift left
  else if (grp == 2) { if (yr > 0)  xx = x[i - 32 * CC]; }   // shift down
  else               { if (yr < 31) xx = x[i + 32 * CC]; }   // shift up
  float a;
  a = mk[c]; xkb[i] = f2bf(fmaf(xv_ - xx, a, xx));
  a = mv[c]; xvb[i] = f2bf(fmaf(xv_ - xx, a, xx));
  a = mr[c]; xrb[i] = f2bf(fmaf(xv_ - xx, a, xx));
}

// ---------------------------------------------------------------------------
// Kernel 2: convert 4 weight matrices (C x C, row-major) f32 -> bf16
// ---------------------------------------------------------------------------
__global__ __launch_bounds__(256) void wconv(
    const float* __restrict__ Wk, const float* __restrict__ Wv,
    const float* __restrict__ Wr, const float* __restrict__ Wo,
    unsigned short* __restrict__ out)
{
  int i = blockIdx.x * 256 + threadIdx.x;          // < C*C
  const float* src = (blockIdx.y == 0) ? Wk :
                     (blockIdx.y == 1) ? Wv :
                     (blockIdx.y == 2) ? Wr : Wo;
  out[(size_t)blockIdx.y * CC * CC + i] = f2bf(src[i]);
}

// ---------------------------------------------------------------------------
// Kernel 3: bf16 WMMA GEMM with TDM-staged weights.
//   Out[m,n] = sum_k A[m,k] * W[n,k]  (+opt sigmoid)
//   block = 8 waves -> 32(M) x 256(N); wave = 16(M) x 64(N)
//   Weight tile (256 rows x 32 k bf16 = 16 KB) is DMA'd into LDS by the
//   Tensor Data Mover, double-buffered: tile i+1 streams in (TENSORcnt)
//   while WMMAs consume tile i from LDS (ds_load_b128).
// ---------------------------------------------------------------------------
__device__ __forceinline__ v16bf load_a_frag(const unsigned short* arow, int kk) {
  // A frag, this code's k order (B matches): lane<16 row=lr holds K {0..15};
  //                      lane>=16 holds {16..31} (hi offset baked in arow)
  v8bf alo = *(const v8bf*)(arow + kk);
  v8bf ahi = *(const v8bf*)(arow + kk + 8);
  return __builtin_shufflevector(alo, ahi,
      0,1,2,3,4,5,6,7,8,9,10,11,12,13,14,15);
}

__global__ __launch_bounds__(256) void gemm_bf16(
    const unsigned short* __restrict__ A,
    const unsigned short* __restrict__ W,
    float* __restrict__ Out, int applySigmoid)
{
  const int K = CC, N = CC;
  // double-buffered weight tile: [buf][row 0..255][k 0..31] bf16
  __shared__ __attribute__((aligned(64))) unsigned short shB[2][256 * 32];

  int lane = threadIdx.x & 31;
  int wid  = threadIdx.x >> 5;                 // 0..7
  int m0   = blockIdx.y * 32 + (wid & 1) * 16;
  int n0blk = blockIdx.x * 256;
  int nloc = (wid >> 1) * 64;                  // wave's col offset inside tile
  int lr = lane & 15;
  int hi = lane >> 4;                          // 0 or 1

  // --- TDM descriptor group 1 (constant across tiles) ---
  // data_size=2B; tensor_dim0=768, tensor_dim1=768; tile 32(k) x 256(rows);
  // tensor_dim0_stride=768
  u32x8 g1;
  g1[0] = 1u << 16;                              // data_size code 1 = 2 bytes
  g1[1] = (768u & 0xFFFFu) << 16;                // tensor_dim0[15:0] @bits63:48
  g1[2] = (768u >> 16) | ((768u & 0xFFFFu) << 16); // dim0[31:16] | dim1[15:0]
  g1[3] = (768u >> 16) | (32u << 16);            // dim1[31:16] | tile_dim0=32
  g1[4] = 256u;                                  // tile_dim1=256, tile_dim2=0
  g1[5] = 768u;                                  // tensor_dim0_stride[31:0]
  g1[6] = 0u;
  g1[7] = 0u;

  auto issue_tile = [&](int kk, int buf) {
    uint64_t gaddr = (uint64_t)(uintptr_t)(W + (size_t)n0blk * K + kk);
    uint32_t laddr = (uint32_t)(uintptr_t)&shB[buf][0];
    u32x4 g0;
    g0[0] = 1u;                                  // count=1 (valid descriptor)
    g0[1] = laddr;                               // lds_addr
    g0[2] = (uint32_t)gaddr;                     // global_addr[31:0]
    g0[3] = (uint32_t)(gaddr >> 32) | (2u << 30);// global_addr[56:32] | type=2
    asm volatile("tensor_load_to_lds %0, %1" :: "s"(g0), "s"(g1) : "memory");
  };

  // prologue: DMA tile 0, prefetch A fragment 0
  if (wid == 0) issue_tile(0, 0);
  const unsigned short* arow = A + (size_t)(m0 + lr) * K + (hi << 4);
  v16bf aC = load_a_frag(arow, 0);
  if (wid == 0) __builtin_amdgcn_s_wait_tensorcnt((short)0);
  __syncthreads();

  v8f acc0 = {}, acc1 = {}, acc2 = {}, acc3 = {};

#pragma unroll
  for (int ik = 0; ik < K / 32; ++ik) {
    int buf = ik & 1;
    // kick off DMA of next tile + prefetch next A fragment (overlaps WMMAs)
    if (ik + 1 < K / 32) {
      if (wid == 0) issue_tile((ik + 1) * 32, buf ^ 1);
    }
    v16bf aN = aC;
    if (ik + 1 < K / 32) aN = load_a_frag(arow, (ik + 1) * 32);

    // B fragments from LDS: col row = nloc + j*16 + lr, k-half = hi
    const char* bbase = (const char*)&shB[buf][0] + hi * 32;
    v16bf b0 = *(const v16bf*)(bbase + (size_t)(nloc +  0 + lr) * 64);
    v16bf b1 = *(const v16bf*)(bbase + (size_t)(nloc + 16 + lr) * 64);
    v16bf b2 = *(const v16bf*)(bbase + (size_t)(nloc + 32 + lr) * 64);
    v16bf b3 = *(const v16bf*)(bbase + (size_t)(nloc + 48 + lr) * 64);
    acc0 = __builtin_amdgcn_wmma_f32_16x16x32_bf16(false, aC, false, b0, (short)0, acc0, false, false);
    acc1 = __builtin_amdgcn_wmma_f32_16x16x32_bf16(false, aC, false, b1, (short)0, acc1, false, false);
    acc2 = __builtin_amdgcn_wmma_f32_16x16x32_bf16(false, aC, false, b2, (short)0, acc2, false, false);
    acc3 = __builtin_amdgcn_wmma_f32_16x16x32_bf16(false, aC, false, b3, (short)0, acc3, false, false);
    aC = aN;

    if (ik + 1 < K / 32) {
      if (wid == 0) __builtin_amdgcn_s_wait_tensorcnt((short)0);
      __syncthreads();   // next tile resident; old buffer free for refill
    }
  }

  // C/D layout: row = vgpr + 8*hi, col = lr
  v8f accs[4] = {acc0, acc1, acc2, acc3};
#pragma unroll
  for (int j = 0; j < 4; ++j) {
    int col = n0blk + nloc + j * 16 + lr;
#pragma unroll
    for (int r = 0; r < 8; ++r) {
      int row = m0 + (hi << 3) + r;
      float vv = accs[j][r];
      if (applySigmoid) vv = 1.f / (1.f + __expf(-vv));
      Out[(size_t)row * N + col] = vv;
    }
  }
}

// ---------------------------------------------------------------------------
// WKV chunked parallel scan (stabilized affine recurrence, associative):
// pass1 chunk summaries -> pass2 sequential prefix of 16 summaries ->
// pass3 re-run chunks from incoming state and emit y.
// ---------------------------------------------------------------------------
__global__ __launch_bounds__(256) void wkv_pass1(
    const float* __restrict__ kbuf, const float* __restrict__ vbuf,
    const float* __restrict__ sd,
    float* __restrict__ sp, float* __restrict__ sq, float* __restrict__ so)
{
  int id  = blockIdx.x * 256 + threadIdx.x;   // < NCH * NCHAN
  int gid = id % NCHAN;
  int ch  = id / NCHAN;
  int c   = gid % CC;
  int bd  = gid / CC;
  int dir = bd & 3;                            // uniform per block
  int b   = bd >> 2;
  float wv = -__expf(sd[dir * CC + c] * (1.0f / TT));
  float p = 0.f, q = 0.f, o = -1e38f;
  const size_t baseKV = (size_t)b * TT * CC + c;
  int s0 = ch * CHL;

  float kt = kbuf[baseKV + (size_t)scan_t(dir, s0) * CC];
  float vt = vbuf[baseKV + (size_t)scan_t(dir, s0) * CC];
#pragma unroll 4
  for (int j = 0; j < CHL; ++j) {
    float ktn = kt, vtn = vt;
    if (j + 1 < CHL) {                         // prefetch next step
      size_t idn = baseKV + (size_t)scan_t(dir, s0 + j + 1) * CC;
      ktn = kbuf[idn]; vtn = vbuf[idn];
    }
    float wo  = wv + o;
    float no2 = fmaxf(wo, kt);
    float A2  = __expf(wo - no2);
    float B2  = __expf(kt - no2);
    p = fmaf(A2, p, B2 * vt);
    q = fmaf(A2, q, B2);
    o = no2;
    kt = ktn; vt = vtn;
  }
  sp[id] = p; sq[id] = q; so[id] = o;
}

__global__ __launch_bounds__(256) void wkv_pass2(
    const float* __restrict__ sd,
    float* __restrict__ sp, float* __restrict__ sq, float* __restrict__ so)
{
  int gid = blockIdx.x * 256 + threadIdx.x;    // < NCHAN
  int c   = gid % CC;
  int dir = (gid / CC) & 3;
  float wv = -__expf(sd[dir * CC + c] * (1.0f / TT));
  float Lw = (float)CHL * wv;
  float p = 0.f, q = 0.f, o = -1e38f;
#pragma unroll
  for (int ch = 0; ch < NCH; ++ch) {
    size_t idx = (size_t)ch * NCHAN + gid;
    float cp = sp[idx], cq = sq[idx], co = so[idx];
    sp[idx] = p; sq[idx] = q; so[idx] = o;     // incoming state for chunk ch
    float os = o + Lw;                          // carry through L decay steps
    float no = fmaxf(os, co);
    float A  = __expf(os - no);
    float Bc = __expf(co - no);
    p = fmaf(A, p, Bc * cp);
    q = fmaf(A, q, Bc * cq);
    o = no;
  }
}

// Scan permutations (H=W=32) are involutions -> scatter at original pixel
// index implements the inverse scan for free.
__global__ __launch_bounds__(256) void wkv_pass3(
    const float* __restrict__ kbuf, const float* __restrict__ vbuf,
    const float* __restrict__ sd,   const float* __restrict__ sf,
    const float* __restrict__ sp, const float* __restrict__ sq,
    const float* __restrict__ so,
    float* __restrict__ ybuf)
{
  int id  = blockIdx.x * 256 + threadIdx.x;   // < NCH * NCHAN
  int gid = id % NCHAN;
  int ch  = id / NCHAN;
  int c   = gid % CC;
  int bd  = gid / CC;
  int dir = bd & 3;
  int b   = bd >> 2;
  int dc  = dir * CC + c;
  float wv = -__expf(sd[dc] * (1.0f / TT));
  float u  = sf[dc] * (1.0f / TT);
  float p = sp[id], q = sq[id], o = so[id];
  const size_t baseKV = (size_t)b * TT * CC + c;
  float* yp = ybuf + (size_t)(dir * BB + b) * TT * CC + c;
  int s0 = ch * CHL;

  int   t  = scan_t(dir, s0);
  float kt = kbuf[baseKV + (size_t)t * CC];
  float vt = vbuf[baseKV + (size_t)t * CC];
#pragma unroll 4
  for (int j = 0; j < CHL; ++j) {
    int tn = t;
    float ktn = kt, vtn = vt;
    if (j + 1 < CHL) {                         // prefetch next step
      tn = scan_t(dir, s0 + j + 1);
      size_t idn = baseKV + (size_t)tn * CC;
      ktn = kbuf[idn]; vtn = vbuf[idn];
    }
    float uk  = u + kt;
    float no  = fmaxf(o, uk);
    float A   = __expf(o - no);
    float Bc  = __expf(uk - no);
    float y   = fmaf(A, p, Bc * vt) / fmaf(A, q, Bc);
    float wo  = wv + o;
    float no2 = fmaxf(wo, kt);
    float A2  = __expf(wo - no2);
    float B2  = __expf(kt - no2);
    p = fmaf(A2, p, B2 * vt);
    q = fmaf(A2, q, B2);
    o = no2;
    yp[(size_t)t * CC] = y;
    t = tn; kt = ktn; vt = vtn;
  }
}

// ---------------------------------------------------------------------------
// Kernel 5: average 4 direction planes, gate with sigmoid(r), emit bf16
// ---------------------------------------------------------------------------
__global__ __launch_bounds__(256) void combine(
    const float* __restrict__ ybuf, const float* __restrict__ sr,
    unsigned short* __restrict__ eos)
{
  size_t i = (size_t)blockIdx.x * 256 + threadIdx.x;   // over B*T*C
  float s = (ybuf[i] + ybuf[i + PL] + ybuf[i + 2 * PL] + ybuf[i + 3 * PL])
            * 0.25f * sr[i];
  eos[i] = f2bf(s);
}

// ---------------------------------------------------------------------------
extern "C" void kernel_launch(void* const* d_in, const int* in_sizes, int n_in,
                              void* d_out, int out_size, void* d_ws, size_t ws_size,
                              hipStream_t stream) {
  const float* x  = (const float*)d_in[0];
  const float* sd = (const float*)d_in[1];
  const float* sf = (const float*)d_in[2];
  const float* mk = (const float*)d_in[3];
  const float* mv = (const float*)d_in[4];
  const float* mr = (const float*)d_in[5];
  const float* Wk = (const float*)d_in[6];
  const float* Wv = (const float*)d_in[7];
  const float* Wr = (const float*)d_in[8];
  const float* Wo = (const float*)d_in[9];
  float* out = (float*)d_out;

  const size_t XB = (size_t)MM * CC * 2;     // bf16 activation plane bytes
  const size_t WB = (size_t)CC * CC * 2;     // bf16 weight matrix bytes
  const size_t FS = (size_t)MM * CC * 4;     // f32 plane bytes
  const size_t SB = (size_t)NCH * NCHAN * 4; // one scan-state array, bytes

  char* ws = (char*)d_ws;
  unsigned short* xkb = (unsigned short*)(ws);
  unsigned short* xvb = (unsigned short*)(ws + XB);
  unsigned short* xrb = (unsigned short*)(ws + 2 * XB);
  unsigned short* wkb = (unsigned short*)(ws + 3 * XB);          // 4 matrices
  float* kbuf = (float*)(ws + 3 * XB + 4 * WB);
  float* vbuf = kbuf + PL;
  float* srb  = vbuf + PL;
  float* ybuf = srb + PL;                                        // 4 planes
  unsigned short* eos = (unsigned short*)(ws + 3 * XB + 4 * WB + 7 * FS);
  float* spb = (float*)(ws + 3 * XB + 4 * WB + 7 * FS + XB);
  float* sqb = (float*)((char*)spb + SB);
  float* sob = (float*)((char*)spb + 2 * SB);

  int nElem = MM * CC;                        // 6 291 456

  // 1) shift + mix -> bf16
  qshift_mix<<<nElem / 256, 256, 0, stream>>>(x, mk, mv, mr, xkb, xvb, xrb);

  // 2) weights -> bf16
  wconv<<<dim3((CC * CC) / 256, 4), 256, 0, stream>>>(Wk, Wv, Wr, Wo, wkb);

  // 3) k/v/r GEMMs (sigmoid fused into r)
  dim3 ggrid(CC / 256, MM / 32);
  gemm_bf16<<<ggrid, 256, 0, stream>>>(xkb, wkb,               kbuf, 0);
  gemm_bf16<<<ggrid, 256, 0, stream>>>(xvb, wkb + CC * CC,     vbuf, 0);
  gemm_bf16<<<ggrid, 256, 0, stream>>>(xrb, wkb + 2 * CC * CC, srb, 1);

  // 4) WKV: chunked parallel scan (summaries -> prefix combine -> emit)
  wkv_pass1<<<(NCH * NCHAN) / 256, 256, 0, stream>>>(kbuf, vbuf, sd, spb, sqb, sob);
  wkv_pass2<<<NCHAN / 256, 256, 0, stream>>>(sd, spb, sqb, sob);
  wkv_pass3<<<(NCH * NCHAN) / 256, 256, 0, stream>>>(kbuf, vbuf, sd, sf,
                                                     spb, sqb, sob, ybuf);

  // 5) average + gate -> bf16
  combine<<<nElem / 256, 256, 0, stream>>>(ybuf, srb, eos);

  // 6) output projection
  gemm_bf16<<<ggrid, 256, 0, stream>>>(eos, wkb + 3 * CC * CC, out, 0);
}
